// MultiHeadAttention_5033701670958
// MI455X (gfx1250) — matrix operand order repaired and checked by reading, not compiled, not run on hardware
//
#include <hip/hip_runtime.h>

typedef _Float16 h16;
typedef __attribute__((ext_vector_type(16))) _Float16 v16h;
typedef __attribute__((ext_vector_type(8)))  _Float16 v8h;
typedef __attribute__((ext_vector_type(4)))  _Float16 v4h;
typedef __attribute__((ext_vector_type(8)))  float    v8f;
typedef __attribute__((ext_vector_type(4)))  float    v4f;

#define WMMA_F16(a, b, c) \
  __builtin_amdgcn_wmma_f32_16x16x32_f16(false, (a), false, (b), (short)0, (c), false, false)

// ---- WMMA fragment loaders (ISA 7.12.2 layouts, wave32), 128-bit loads ----
// A: 16x32 (MxK), row-major source, leading dim ld (multiple of 8).
// lane<16: M=lane, K=0..7 & 16..23 ; lane>=16: M=lane-16, K=8..15 & 24..31
static __device__ __forceinline__ v16h load_a_rm(const h16* p, int ld) {
  const int lane = threadIdx.x & 31;
  const h16* r = p + (lane & 15) * ld + ((lane >> 4) << 3);
  const v8h lo = *(const v8h*)r;        // 16B
  const v8h hi = *(const v8h*)(r + 16); // 16B
  return __builtin_shufflevector(lo, hi, 0, 1, 2, 3, 4, 5, 6, 7,
                                         8, 9, 10, 11, 12, 13, 14, 15);
}
// B: 32x16 (KxN) from N-major source (rows = N, contiguous K), leading dim ld.
// lane<16: N=lane, K=0..7 & 16..23 ; lane>=16: N=lane-16, K=8..15 & 24..31
static __device__ __forceinline__ v16h load_b_nk(const h16* p, int ld) {
  const int lane = threadIdx.x & 31;
  const h16* r = p + (lane & 15) * ld + ((lane >> 4) << 3);
  const v8h lo = *(const v8h*)r;        // 16B
  const v8h hi = *(const v8h*)(r + 16); // 16B
  return __builtin_shufflevector(lo, hi, 0, 1, 2, 3, 4, 5, 6, 7,
                                         8, 9, 10, 11, 12, 13, 14, 15);
}

// ===================== QKV projection: Y = X @ W + bias (f16 out) =========
// X: 4096x512 fp32, W: 512x512 fp32. B tile staged TRANSPOSED (N-major) in LDS
// so B-fragments are contiguous 128-bit loads.
// vtrans==0: dst[((b*8+h)*2048+s)*64+dh]  (row-major per head)
// vtrans==1: dst[((b*8+h)*64+dh)*2048+s]  (dh-major per head, for attn@V B-frags)
__global__ __launch_bounds__(256) void proj_kernel(
    const float* __restrict__ X, const float* __restrict__ W,
    const float* __restrict__ bias, h16* __restrict__ dst, int vtrans) {
  __shared__ __align__(16) h16 As[64 * 32];   // 64 M x 32 K, row-major
  __shared__ __align__(16) h16 BsT[64 * 32];  // 64 N x 32 K, N-major
  const int tid = threadIdx.x, wave = tid >> 5, lane = tid & 31;
  const int wr = wave >> 1, wc = wave & 1;      // 4x2 wave grid, 16x32 per wave
  const int m0 = blockIdx.y << 6, n0 = blockIdx.x << 6;
  const int ar = tid >> 2, ac = (tid & 3) << 3; // A: 8 contiguous fp32/thread
  const int br = tid >> 3, bc = (tid & 7) << 3; // B: 8 contiguous fp32/thread
  v8f acc0 = {}, acc1 = {};
  for (int kk = 0; kk < 512; kk += 32) {
    const float* xp = X + (size_t)(m0 + ar) * 512 + kk + ac;
    const float* wp = W + (size_t)(kk + br) * 512 + n0 + bc;
    const v4f x0 = *(const v4f*)xp, x1 = *(const v4f*)(xp + 4);
    const v4f w0 = *(const v4f*)wp, w1 = *(const v4f*)(wp + 4);
    v8h ah;
#pragma unroll
    for (int j = 0; j < 4; ++j) { ah[j] = (h16)x0[j]; ah[4 + j] = (h16)x1[j]; }
    *(v8h*)(As + ar * 32 + ac) = ah;            // one 16B LDS store
#pragma unroll
    for (int j = 0; j < 4; ++j) {               // transposed (strided) stores
      BsT[(bc + j) * 32 + br] = (h16)w0[j];
      BsT[(bc + 4 + j) * 32 + br] = (h16)w1[j];
    }
    __syncthreads();
    v16h a  = load_a_rm(As + wr * 16 * 32, 32);
    v16h b0 = load_b_nk(BsT + (wc * 32) * 32, 32);
    v16h b1 = load_b_nk(BsT + (wc * 32 + 16) * 32, 32);
    acc0 = WMMA_F16(a, b0, acc0);
    acc1 = WMMA_F16(a, b1, acc1);
    __syncthreads();
  }
  const int row = (lane >> 4) << 3, col = lane & 15;
#pragma unroll
  for (int t = 0; t < 2; ++t) {
    v8f& acc = t ? acc1 : acc0;
#pragma unroll
    for (int v = 0; v < 8; ++v) {
      const int m = m0 + wr * 16 + row + v;
      const int n = n0 + wc * 32 + t * 16 + col;
      const float val = acc[v] + bias[n];
      const int b = m >> 11, s = m & 2047, h = n >> 6, dh = n & 63;
      const long o = vtrans ? ((long)((b * 8 + h) * 64 + dh) << 11) + s
                            : (((long)(b * 8 + h) * 2048 + s) << 6) + dh;
      dst[o] = (h16)val;
    }
  }
}

// ===================== fp32 -> f16 convert (for E) ========================
__global__ void cvt_kernel(const float* __restrict__ src, h16* __restrict__ dst, int n) {
  const int i = blockIdx.x * blockDim.x + threadIdx.x;
  if (i < n) dst[i] = (h16)src[i];
}

// ===================== Fused attention ====================================
// One block = 16 query rows for one (b,h):
//  logits[16][2048] (fp32, LDS) = Q K^T ; += shifted QE (Srel) ; softmax ;
//  attn -> d_out (fp32, once, 128-bit stores) + f16 strip in LDS ; O = attn @ V.
__global__ __launch_bounds__(256) void attn_kernel(
    const h16* __restrict__ Qh, const h16* __restrict__ Kh,
    const h16* __restrict__ Vt, const h16* __restrict__ Ef,
    float* __restrict__ attn_out, h16* __restrict__ Oh) {
  __shared__ __align__(16) unsigned char smem[16 * 2048 * 4 + 16 * 2048 * 2]; // 192 KB
  float* logits = (float*)smem;                 // 16 x 2048 fp32
  h16*   attnh  = (h16*)(smem + 16 * 2048 * 4); // 16 x 2048 f16
  float* oacc   = (float*)smem;                 // 16 x 64 fp32 (aliases dead logits)

  const int tid = threadIdx.x, wave = tid >> 5, lane = tid & 31;
  const int qb = blockIdx.x;   // 0..127
  const int bh = blockIdx.y;   // 0..15
  const int q0 = qb << 4;

  const h16* Qp = Qh + ((long)bh * 2048 + q0) * 64;
  const h16* Kp = Kh + (long)bh * 2048 * 64;
  const h16* Vp = Vt + (long)bh * 64 * 2048;

  // Q fragments (16 rows x 64 = two K-steps), loaded straight from global
  const v16h aq0 = load_a_rm(Qp, 64);
  const v16h aq1 = load_a_rm(Qp + 32, 64);
  const int r0 = (lane >> 4) << 3, col = lane & 15;

  // ---- Phase 1: logits = Q K^T over all 128 k-tiles (waves stride) ----
  for (int kt = wave; kt < 128; kt += 8) {
    const h16* kp = Kp + ((long)kt << 4) * 64;
    if (kt + 8 < 128) __builtin_prefetch(Kp + ((long)(kt + 8) << 4) * 64, 0, 1);
    v16h b0 = load_b_nk(kp, 64);
    v16h b1 = load_b_nk(kp + 32, 64);
    v8f c = {};
    c = WMMA_F16(aq0, b0, c);
    c = WMMA_F16(aq1, b1, c);
#pragma unroll
    for (int v = 0; v < 8; ++v)
      logits[(r0 + v) * 2048 + (kt << 4) + col] = c[v];
  }
  __syncthreads();

  // ---- Phase 2: Srel[q][k] = Q[q] . E[2047-(q-k)]  (k<=q) ----
  // QE[q][j] lands at k = j - 2047 + q : per-row shifted add, j -> k injective,
  // wave-disjoint j-tiles => race-free non-atomic adds.
  for (int jt = 127 - qb + wave; jt < 128; jt += 8) {
    const h16* ep = Ef + ((long)jt << 4) * 64;
    v16h b0 = load_b_nk(ep, 64);
    v16h b1 = load_b_nk(ep + 32, 64);
    v8f c = {};
    c = WMMA_F16(aq0, b0, c);
    c = WMMA_F16(aq1, b1, c);
#pragma unroll
    for (int v = 0; v < 8; ++v) {
      const int r = r0 + v;
      const int k = (jt << 4) + col - 2047 + (q0 + r);
      if (k >= 0) logits[r * 2048 + k] += c[v];
    }
  }
  __syncthreads();

  // ---- Phase 3: scale + causal mask + softmax (vectorized x4) ----
  for (int r = wave; r < 16; r += 8) {  // 2 rows per wave
    const int qg = q0 + r;
    float* lrow = logits + r * 2048;
    float mx = -3.4e38f;
    for (int k = lane << 2; k < 2048; k += 128) {
      v4f x = *(v4f*)(lrow + k);
#pragma unroll
      for (int j = 0; j < 4; ++j) {
        float xv = x[j] * 0.125f;          // 1/sqrt(DH)
        if (k + j > qg) xv = -1.0e9f;      // causal mask
        x[j] = xv;
        mx = fmaxf(mx, xv);
      }
      *(v4f*)(lrow + k) = x;
    }
#pragma unroll
    for (int d = 16; d; d >>= 1) mx = fmaxf(mx, __shfl_xor(mx, d, 32));
    float sum = 0.f;
    for (int k = lane << 2; k < 2048; k += 128) {
      v4f x = *(v4f*)(lrow + k);
#pragma unroll
      for (int j = 0; j < 4; ++j) {
        const float p = __expf(x[j] - mx);
        x[j] = p;
        sum += p;
      }
      *(v4f*)(lrow + k) = x;
    }
#pragma unroll
    for (int d = 16; d; d >>= 1) sum += __shfl_xor(sum, d, 32);
    const float inv = 1.0f / sum;
    float* arow = attn_out + ((long)bh * 2048 + qg) * 2048;
    for (int k = lane << 2; k < 2048; k += 128) {
      v4f x = *(v4f*)(lrow + k);
      v4h ph;
#pragma unroll
      for (int j = 0; j < 4; ++j) {
        const float p = x[j] * inv;
        x[j] = p;
        ph[j] = (h16)p;
      }
      *(v4f*)(arow + k) = x;               // 128-bit HBM store
      *(v4h*)(attnh + r * 2048 + k) = ph;  // 64-bit LDS store
    }
  }
  __syncthreads();

  // ---- Phase 4: O(16x64) = attn(16x2048) @ V(2048x64), K split by wave ----
  for (int i = tid; i < 16 * 64; i += 256) oacc[i] = 0.f;  // logits now dead
  __syncthreads();
  v8f co[4] = {};
  const int ks0 = wave << 8;                 // 256 K per wave
  for (int ks = ks0; ks < ks0 + 256; ks += 32) {
    v16h a = load_a_rm(attnh + ks, 2048);
#pragma unroll
    for (int n = 0; n < 4; ++n) {
      v16h b = load_b_nk(Vp + ((long)(n << 4)) * 2048 + ks, 2048);
      co[n] = WMMA_F16(a, b, co[n]);
    }
  }
#pragma unroll
  for (int n = 0; n < 4; ++n)
#pragma unroll
    for (int v = 0; v < 8; ++v)
      atomicAdd(&oacc[(r0 + v) * 64 + (n << 4) + col], co[n][v]);
  __syncthreads();
  const int b = bh >> 3, h = bh & 7;
  for (int i = tid; i < 16 * 64; i += 256) {
    const int r = i >> 6, dh = i & 63;
    const int s = q0 + r;
    Oh[(((long)(b * 2048 + s) * 8 + h) << 6) + dh] = (h16)oacc[i];
  }
}

// ===================== Output projection: out = Oh @ Wo + bo (fp32) =======
__global__ __launch_bounds__(256) void outproj_kernel(
    const h16* __restrict__ X, const float* __restrict__ W,
    const float* __restrict__ bias, float* __restrict__ out) {
  __shared__ __align__(16) h16 As[64 * 32];
  __shared__ __align__(16) h16 BsT[64 * 32];  // N-major
  const int tid = threadIdx.x, wave = tid >> 5, lane = tid & 31;
  const int wr = wave >> 1, wc = wave & 1;
  const int m0 = blockIdx.y << 6, n0 = blockIdx.x << 6;
  const int ar = tid >> 2, ac = (tid & 3) << 3;
  const int br = tid >> 3, bc = (tid & 7) << 3;
  v8f acc0 = {}, acc1 = {};
  for (int kk = 0; kk < 512; kk += 32) {
    const h16*   xp = X + (size_t)(m0 + ar) * 512 + kk + ac;
    const float* wp = W + (size_t)(kk + br) * 512 + n0 + bc;
    *(v8h*)(As + ar * 32 + ac) = *(const v8h*)xp;  // 16B copy
    const v4f w0 = *(const v4f*)wp, w1 = *(const v4f*)(wp + 4);
#pragma unroll
    for (int j = 0; j < 4; ++j) {
      BsT[(bc + j) * 32 + br] = (h16)w0[j];
      BsT[(bc + 4 + j) * 32 + br] = (h16)w1[j];
    }
    __syncthreads();
    v16h a  = load_a_rm(As + wr * 16 * 32, 32);
    v16h b0 = load_b_nk(BsT + (wc * 32) * 32, 32);
    v16h b1 = load_b_nk(BsT + (wc * 32 + 16) * 32, 32);
    acc0 = WMMA_F16(a, b0, acc0);
    acc1 = WMMA_F16(a, b1, acc1);
    __syncthreads();
  }
  const int row = (lane >> 4) << 3, col = lane & 15;
#pragma unroll
  for (int t = 0; t < 2; ++t) {
    v8f& acc = t ? acc1 : acc0;
#pragma unroll
    for (int v = 0; v < 8; ++v) {
      const int m = m0 + wr * 16 + row + v;
      const int n = n0 + wc * 32 + t * 16 + col;
      out[(size_t)m * 512 + n] = acc[v] + bias[n];
    }
  }
}

// ===================== launch ============================================
extern "C" void kernel_launch(void* const* d_in, const int* in_sizes, int n_in,
                              void* d_out, int out_size, void* d_ws, size_t ws_size,
                              hipStream_t stream) {
  const float* v  = (const float*)d_in[0];
  const float* k  = (const float*)d_in[1];
  const float* q  = (const float*)d_in[2];
  // d_in[3] = mask (implemented analytically)
  const float* Wq = (const float*)d_in[4];
  const float* bq = (const float*)d_in[5];
  const float* Wk = (const float*)d_in[6];
  const float* bk = (const float*)d_in[7];
  const float* Wv = (const float*)d_in[8];
  const float* bv = (const float*)d_in[9];
  const float* Wo = (const float*)d_in[10];
  const float* bo = (const float*)d_in[11];
  const float* E  = (const float*)d_in[12];

  const size_t HEADSZ = (size_t)2 * 8 * 2048 * 64;  // B*H*S*DH
  h16* Qh = (h16*)d_ws;
  h16* Kh = Qh + HEADSZ;
  h16* Vt = Kh + HEADSZ;
  h16* Ef = Vt + HEADSZ;
  h16* Oh = Ef + (size_t)2048 * 64;

  float* out  = (float*)d_out;
  float* attn = out + (size_t)2 * 2048 * 512;

  const dim3 g1(8, 64), b256(256);
  proj_kernel<<<g1, b256, 0, stream>>>(q, Wq, bq, Qh, 0);
  proj_kernel<<<g1, b256, 0, stream>>>(k, Wk, bk, Kh, 0);
  proj_kernel<<<g1, b256, 0, stream>>>(v, Wv, bv, Vt, 1);
  cvt_kernel<<<(2048 * 64 + 255) / 256, b256, 0, stream>>>(E, Ef, 2048 * 64);
  attn_kernel<<<dim3(128, 16), b256, 0, stream>>>(Qh, Kh, Vt, Ef, attn, Oh);
  outproj_kernel<<<g1, b256, 0, stream>>>(Oh, Wo, bo, out);
}
